// CifarNetMem_20993800142979
// MI455X (gfx1250) — matrix operand order repaired and checked by reading, not compiled, not run on hardware
//
#include <hip/hip_runtime.h>
#include <hip/hip_bf16.h>

// CIFAR CNN forward on gfx1250 via FP16 WMMA implicit GEMM.
// LUT = linspace(-4,4,129) => quantize = round to nearest 1/16, clamp +-4
// (ties to lower level). All quantized values are exact in FP16.
// Workspace requirement: ~115 MB (activations NHWC f16 + packed weights).

typedef __attribute__((ext_vector_type(16))) _Float16 v16h;
typedef __attribute__((ext_vector_type(8)))  _Float16 v8h;
typedef __attribute__((ext_vector_type(8)))  float    v8f;

__device__ __forceinline__ float lutq(float v) {
    // nearest multiple of 1/16, ties toward lower value, clamp to [-4,4]
    float t = ceilf(v * 16.0f - 0.5f);
    t = fminf(fmaxf(t, -64.0f), 64.0f);
    return t * 0.0625f;
}

// ---------------------------------------------------------------------------
// Input: fp32 NCHW [512,3,32,32] -> f16 NHWC [512,32,32,32] (C padded w/ 0)
// ---------------------------------------------------------------------------
__global__ __launch_bounds__(256) void k_input_pack(const float* __restrict__ x,
                                                    _Float16* __restrict__ out) {
    int id = blockIdx.x * blockDim.x + threadIdx.x;   // one thread per pixel
    if (id >= 512 * 1024) return;
    int b = id >> 10;
    int p = id & 1023;
    v8h z = {};
    v8h first = z;
    first[0] = (_Float16)x[(size_t)b * 3072 + 0 * 1024 + p];
    first[1] = (_Float16)x[(size_t)b * 3072 + 1 * 1024 + p];
    first[2] = (_Float16)x[(size_t)b * 3072 + 2 * 1024 + p];
    _Float16* o = out + (size_t)id * 32;
    *(v8h*)(o + 0)  = first;
    *(v8h*)(o + 8)  = z;
    *(v8h*)(o + 16) = z;
    *(v8h*)(o + 24) = z;
}

// ---------------------------------------------------------------------------
// A-fragment index mapping used here (16-bit A 16x32, same K order as B, not the ISA table):
// lane 0-15 hold M=lane, element j -> K=(lane/16)*16+j (contiguous)
// Packed layout: apack[((kc*Mt + mt)*32 + lane)*16 + j]
// ---------------------------------------------------------------------------
__device__ __forceinline__ int a_k_of_j(int lane, int j) {
    int i = j >> 1, lo = j & 1, kb = lane >> 4;
    (void)i; (void)lo;
    return kb * 16 + j;
}

// Conv weights [Co][Cin][3][3] fp32 -> quantized f16 A-fragments,
// K order = (ky*3+kx)*Cinp + ci   (ci >= Cin zero padded)
__global__ __launch_bounds__(256) void k_pack_conv(const float* __restrict__ w,
                                                   _Float16* __restrict__ apack,
                                                   int Co, int Cin, int Cinp) {
    int id = blockIdx.x * blockDim.x + threadIdx.x;
    int total = 9 * Cinp * Co;
    if (id >= total) return;
    int j = id & 15, lane = (id >> 4) & 31, rest = id >> 9;
    int Mt = Co >> 4;
    int mt = rest % Mt, kc = rest / Mt;
    int m = mt * 16 + (lane & 15);
    int kin = a_k_of_j(lane, j);
    int cpc = Cinp >> 5;
    int kyx = kc / cpc, cc = kc - kyx * cpc;
    int ky = kyx / 3, kx = kyx - ky * 3;
    int ci = cc * 32 + kin;
    float v = 0.0f;
    if (ci < Cin) v = lutq(w[(((size_t)m * Cin + ci) * 3 + ky) * 3 + kx]);
    apack[id] = (_Float16)v;
}

// FC weights [Mreal][K] fp32 -> f16 A-fragments, M padded to Mpad.
// perm==1 (FC1): our feature k = (h*4+w)*128 + c  ->  ref k = c*16 + (h*4+w)
__global__ __launch_bounds__(256) void k_pack_fc(const float* __restrict__ w,
                                                 _Float16* __restrict__ apack,
                                                 int Mreal, int Mpad, int K, int perm) {
    int id = blockIdx.x * blockDim.x + threadIdx.x;
    int total = K * Mpad;
    if (id >= total) return;
    int j = id & 15, lane = (id >> 4) & 31, rest = id >> 9;
    int Mt = Mpad >> 4;
    int mt = rest % Mt, kc = rest / Mt;
    int m = mt * 16 + (lane & 15);
    int k = kc * 32 + a_k_of_j(lane, j);
    float v = 0.0f;
    if (m < Mreal) {
        int kr = perm ? ((k & 127) * 16 + (k >> 7)) : k;
        v = lutq(w[(size_t)m * K + kr]);
    }
    apack[id] = (_Float16)v;
}

// ---------------------------------------------------------------------------
// Implicit-GEMM 3x3 SAME conv.
// Wave computes 32 Cout (two 16-wide M tiles sharing each B fragment)
// x 16 pixels. act NHWC f16 [B][H][W][Cinp], H == W == (1<<lw).
// Epilogue: +q(bias), ReLU, LUT-quant, packed f16 stores.
// ---------------------------------------------------------------------------
__global__ __launch_bounds__(256) void k_conv_wmma(const _Float16* __restrict__ in,
                                                   const _Float16* __restrict__ apack,
                                                   const float* __restrict__ bias,
                                                   _Float16* __restrict__ out,
                                                   int Bn, int H, int W, int lw,
                                                   int Cinp, int Cout) {
    const int lane = threadIdx.x & 31;
    const int wave = (blockIdx.x * blockDim.x + threadIdx.x) >> 5;
    const int Mt  = Cout >> 4;
    const int Mt2 = Mt >> 1;             // M-tile pairs per pixel tile
    const int npix = H * W;
    const int ntN = npix >> 4;
    const int total = Bn * ntN * Mt2;
    if (wave >= total) return;
    int mtp = wave % Mt2;
    int pt  = (wave / Mt2) % ntN;
    int b   = wave / (Mt2 * ntN);
    const int mt0 = mtp * 2;

    const int n  = lane & 15;            // output pixel within tile (B/D column)
    const int kb = lane >> 4;            // which K half this lane carries
    const int p = pt * 16 + n;
    const int y = p >> lw;
    const int x = p & (W - 1);

    const int cpc = Cinp >> 5;           // 32-channel chunks per tap
    const size_t astride = (size_t)Mt * 512;

    v8f acc0 = {}, acc1 = {};
    const _Float16* aptr  = apack + (size_t)mt0 * 512 + (size_t)lane * 16;
    const _Float16* ibase = in + (size_t)b * npix * Cinp + (size_t)kb * 16;

    for (int ky = 0; ky < 3; ++ky) {
        const int ys = y + ky - 1;
        const bool vy = (unsigned)ys < (unsigned)H;
        // warm next row's cachelines in the WGP cache (global_prefetch_b8)
        if ((unsigned)(ys + 1) < (unsigned)H)
            __builtin_prefetch(ibase + (size_t)((ys + 1) * W + x) * Cinp, 0, 3);
        for (int kx = 0; kx < 3; ++kx) {
            const int xs = x + kx - 1;
            const bool v = vy && ((unsigned)xs < (unsigned)W);
            const _Float16* brow = ibase + (size_t)(ys * W + xs) * Cinp;
            for (int cc = 0; cc < cpc; ++cc) {
                v16h bf = {};
                if (v) bf = *(const v16h*)(brow + cc * 32);
                v16h a0 = *(const v16h*)(aptr);
                v16h a1 = *(const v16h*)(aptr + 512);
                acc0 = __builtin_amdgcn_wmma_f32_16x16x32_f16(false, a0, false, bf,
                                                              (short)0, acc0, false, false);
                acc1 = __builtin_amdgcn_wmma_f32_16x16x32_f16(false, a1, false, bf,
                                                              (short)0, acc1, false, false);
                aptr += astride;
            }
        }
    }

    // D layout: vreg r holds M = r + 8*kb, column = n  -> 8 contiguous channels
    const int co0 = mt0 * 16 + kb * 8;
    _Float16* obase = out + ((size_t)b * npix + p) * Cout;
    v8h hv0, hv1;
#pragma unroll
    for (int r = 0; r < 8; ++r) {
        float v0 = acc0[r] + lutq(bias[co0 + r]);
        float v1 = acc1[r] + lutq(bias[co0 + 16 + r]);
        hv0[r] = (_Float16)lutq(fmaxf(v0, 0.0f));
        hv1[r] = (_Float16)lutq(fmaxf(v1, 0.0f));
    }
    *(v8h*)(obase + co0)      = hv0;
    *(v8h*)(obase + co0 + 16) = hv1;
}

// ---------------------------------------------------------------------------
// 2x2 max pool, NHWC f16, 8 channels per thread (b128 loads/stores)
// ---------------------------------------------------------------------------
__global__ __launch_bounds__(256) void k_pool(const _Float16* __restrict__ in,
                                              _Float16* __restrict__ out,
                                              int Bn, int H, int W, int C) {
    int Ho = H >> 1, Wo = W >> 1, cv = C >> 3;
    size_t id = (size_t)blockIdx.x * blockDim.x + threadIdx.x;
    size_t tot = (size_t)Bn * Ho * Wo * cv;
    if (id >= tot) return;
    int c8 = (int)(id % cv); size_t r = id / cv;
    int xo = (int)(r % Wo); r /= Wo;
    int yo = (int)(r % Ho); int b = (int)(r / Ho);
    const _Float16* p00 = in + (((size_t)b * H + yo * 2) * W + xo * 2) * C + c8 * 8;
    v8h a = *(const v8h*)p00;
    v8h bb = *(const v8h*)(p00 + C);
    v8h c = *(const v8h*)(p00 + (size_t)W * C);
    v8h d = *(const v8h*)(p00 + (size_t)W * C + C);
    v8h m;
#pragma unroll
    for (int i = 0; i < 8; ++i) {
        _Float16 t0 = a[i] > bb[i] ? a[i] : bb[i];
        _Float16 t1 = c[i] > d[i] ? c[i] : d[i];
        m[i] = t0 > t1 ? t0 : t1;
    }
    *(v8h*)(out + (((size_t)b * Ho + yo) * Wo + xo) * C + c8 * 8) = m;
}

// ---------------------------------------------------------------------------
// FC GEMM: wave = 16 couts x 16 batch samples. in [n][K] f16.
// mode 0: relu+quant -> f16 out (stride=Mpad). mode 1: bias only -> fp32 out.
// ---------------------------------------------------------------------------
__global__ __launch_bounds__(256) void k_fc_wmma(const _Float16* __restrict__ in,
                                                 const _Float16* __restrict__ apack,
                                                 const float* __restrict__ bias,
                                                 _Float16* __restrict__ out16,
                                                 float* __restrict__ out32,
                                                 int N, int K, int Mpad, int Mreal,
                                                 int outStride, int mode) {
    const int lane = threadIdx.x & 31;
    const int wave = (blockIdx.x * blockDim.x + threadIdx.x) >> 5;
    const int Mt = Mpad >> 4;
    const int total = Mt * (N >> 4);
    if (wave >= total) return;
    int mt = wave % Mt;
    int nt = wave / Mt;
    const int n  = lane & 15;
    const int kb = lane >> 4;
    const int sample = nt * 16 + n;

    const _Float16* brow = in + (size_t)sample * K + kb * 16;
    const _Float16* abase = apack + (size_t)mt * 512 + (size_t)lane * 16;
    const size_t astride = (size_t)Mt * 512;
    const int nkc = K >> 5;

    v8f acc = {};
#pragma unroll 4
    for (int kc = 0; kc < nkc; ++kc) {
        v16h bf = *(const v16h*)(brow + (size_t)kc * 32);
        v16h af = *(const v16h*)(abase + (size_t)kc * astride);
        acc = __builtin_amdgcn_wmma_f32_16x16x32_f16(false, af, false, bf,
                                                     (short)0, acc, false, false);
    }

    const int co0 = mt * 16 + kb * 8;
    if (mode == 0) {
        v8h hv;
#pragma unroll
        for (int r = 0; r < 8; ++r) {
            float v = acc[r] + lutq(bias[co0 + r]);
            v = fmaxf(v, 0.0f);
            hv[r] = (_Float16)lutq(v);
        }
        *(v8h*)(out16 + (size_t)sample * outStride + co0) = hv;
    } else {
#pragma unroll
        for (int r = 0; r < 8; ++r) {
            int m = co0 + r;
            if (m < Mreal) {
                out32[(size_t)sample * outStride + m] = acc[r] + lutq(bias[m]);
            }
        }
    }
}

// ---------------------------------------------------------------------------
// Softmax over 10 logits per row (logits stride 16 fp32)
// ---------------------------------------------------------------------------
__global__ __launch_bounds__(256) void k_softmax(const float* __restrict__ logits,
                                                 float* __restrict__ out) {
    int r = blockIdx.x * blockDim.x + threadIdx.x;
    if (r >= 512) return;
    const float* l = logits + (size_t)r * 16;
    float mx = l[0];
#pragma unroll
    for (int c = 1; c < 10; ++c) mx = fmaxf(mx, l[c]);
    float e[10], s = 0.0f;
#pragma unroll
    for (int c = 0; c < 10; ++c) { e[c] = expf(l[c] - mx); s += e[c]; }
    float inv = 1.0f / s;
#pragma unroll
    for (int c = 0; c < 10; ++c) out[(size_t)r * 10 + c] = e[c] * inv;
}

// ---------------------------------------------------------------------------
extern "C" void kernel_launch(void* const* d_in, const int* in_sizes, int n_in,
                              void* d_out, int out_size, void* d_ws, size_t ws_size,
                              hipStream_t stream) {
    (void)in_sizes; (void)n_in; (void)out_size; (void)ws_size;
    const float* x   = (const float*)d_in[0];
    // d_in[1] = lut (linspace(-4,4,129)); hardcoded as step 1/16, clamp +-4
    const float* w1  = (const float*)d_in[2];  const float* b1  = (const float*)d_in[3];
    const float* w2  = (const float*)d_in[4];  const float* b2  = (const float*)d_in[5];
    const float* w3  = (const float*)d_in[6];  const float* b3  = (const float*)d_in[7];
    const float* w4  = (const float*)d_in[8];  const float* b4  = (const float*)d_in[9];
    const float* w5  = (const float*)d_in[10]; const float* b5  = (const float*)d_in[11];
    const float* w6  = (const float*)d_in[12]; const float* b6  = (const float*)d_in[13];
    const float* fw1 = (const float*)d_in[14]; const float* fb1 = (const float*)d_in[15];
    const float* fw2 = (const float*)d_in[16]; const float* fb2 = (const float*)d_in[17];
    const float* fw3 = (const float*)d_in[18]; const float* fb3 = (const float*)d_in[19];

    char* ws = (char*)d_ws;
    size_t off = 0;
    auto alloc = [&](size_t bytes) -> char* {
        char* p = ws + off;
        off = (off + bytes + 255) & ~(size_t)255;
        return p;
    };
    _Float16* X16 = (_Float16*)alloc(512ull * 1024 * 32 * 2);  // input NHWC pad32
    _Float16* A   = (_Float16*)alloc(512ull * 1024 * 32 * 2);  // ping
    _Float16* B   = (_Float16*)alloc(512ull * 1024 * 32 * 2);  // pong
    _Float16* P   = (_Float16*)alloc(512ull * 256 * 32 * 2);   // pooled
    _Float16* F1  = (_Float16*)alloc(512ull * 512 * 2);
    _Float16* F2  = (_Float16*)alloc(512ull * 128 * 2);
    float*    LOG = (float*)   alloc(512ull * 16 * 4);
    _Float16* PC1 = (_Float16*)alloc(9ull * 32 * 32 * 2);
    _Float16* PC2 = (_Float16*)alloc(9ull * 32 * 32 * 2);
    _Float16* PC3 = (_Float16*)alloc(9ull * 32 * 64 * 2);
    _Float16* PC4 = (_Float16*)alloc(9ull * 64 * 64 * 2);
    _Float16* PC5 = (_Float16*)alloc(9ull * 64 * 128 * 2);
    _Float16* PC6 = (_Float16*)alloc(9ull * 128 * 128 * 2);
    _Float16* PF1 = (_Float16*)alloc(2048ull * 512 * 2);
    _Float16* PF2 = (_Float16*)alloc(512ull * 128 * 2);
    _Float16* PF3 = (_Float16*)alloc(128ull * 16 * 2);

    auto blk = [](long long threads) { return (unsigned)((threads + 255) / 256); };

    // --- prep ---
    k_input_pack<<<blk(512 * 1024), 256, 0, stream>>>(x, X16);
    k_pack_conv<<<blk(9 * 32 * 32), 256, 0, stream>>>(w1, PC1, 32, 3, 32);
    k_pack_conv<<<blk(9 * 32 * 32), 256, 0, stream>>>(w2, PC2, 32, 32, 32);
    k_pack_conv<<<blk(9 * 32 * 64), 256, 0, stream>>>(w3, PC3, 64, 32, 32);
    k_pack_conv<<<blk(9 * 64 * 64), 256, 0, stream>>>(w4, PC4, 64, 64, 64);
    k_pack_conv<<<blk(9 * 64 * 128), 256, 0, stream>>>(w5, PC5, 128, 64, 64);
    k_pack_conv<<<blk(9 * 128 * 128), 256, 0, stream>>>(w6, PC6, 128, 128, 128);
    k_pack_fc<<<blk(2048 * 512), 256, 0, stream>>>(fw1, PF1, 512, 512, 2048, 1);
    k_pack_fc<<<blk(512 * 128), 256, 0, stream>>>(fw2, PF2, 128, 128, 512, 0);
    k_pack_fc<<<blk(128 * 16), 256, 0, stream>>>(fw3, PF3, 10, 16, 128, 0);

    // --- conv stack: waves = B * (HW/16) * (Cout/32), 32 threads/wave ---
    k_conv_wmma<<<blk(32768ll * 32), 256, 0, stream>>>(X16, PC1, b1, A, 512, 32, 32, 5, 32, 32);
    k_conv_wmma<<<blk(32768ll * 32), 256, 0, stream>>>(A, PC2, b2, B, 512, 32, 32, 5, 32, 32);
    k_pool<<<blk(512ll * 16 * 16 * 4), 256, 0, stream>>>(B, P, 512, 32, 32, 32);
    k_conv_wmma<<<blk(16384ll * 32), 256, 0, stream>>>(P, PC3, b3, A, 512, 16, 16, 4, 32, 64);
    k_conv_wmma<<<blk(16384ll * 32), 256, 0, stream>>>(A, PC4, b4, B, 512, 16, 16, 4, 64, 64);
    k_pool<<<blk(512ll * 8 * 8 * 8), 256, 0, stream>>>(B, P, 512, 16, 16, 64);
    k_conv_wmma<<<blk(8192ll * 32), 256, 0, stream>>>(P, PC5, b5, A, 512, 8, 8, 3, 64, 128);
    k_conv_wmma<<<blk(8192ll * 32), 256, 0, stream>>>(A, PC6, b6, B, 512, 8, 8, 3, 128, 128);
    k_pool<<<blk(512ll * 4 * 4 * 16), 256, 0, stream>>>(B, P, 512, 8, 8, 128);

    // --- FC stack ---
    k_fc_wmma<<<blk(1024ll * 32), 256, 0, stream>>>(P, PF1, fb1, F1, nullptr,
                                                    512, 2048, 512, 512, 512, 0);
    k_fc_wmma<<<blk(256ll * 32), 256, 0, stream>>>(F1, PF2, fb2, F2, nullptr,
                                                   512, 512, 128, 128, 128, 0);
    k_fc_wmma<<<blk(32ll * 32), 256, 0, stream>>>(F2, PF3, fb3, nullptr, LOG,
                                                  512, 128, 16, 10, 16, 1);
    k_softmax<<<blk(512), 256, 0, stream>>>(LOG, (float*)d_out);
}
